// DGCNN_sim_32633161515573
// MI455X (gfx1250) — matrix operand order repaired and checked by reading, not compiled, not run on hardware
//
#include <hip/hip_runtime.h>
#include <hip/hip_bf16.h>

#define KNN 20
#define NEG_SLOPE 0.2f

typedef __attribute__((ext_vector_type(16))) _Float16 v16h;
typedef __attribute__((ext_vector_type(8)))  _Float16 v8h;
typedef __attribute__((ext_vector_type(8)))  float    v8f;
typedef __attribute__((ext_vector_type(4)))  int      v4i;

static constexpr int NB   = 8;     // batch
static constexpr int NPTS = 4096;  // points

__device__ __forceinline__ float lrelu_f(float v) {
    return v >= 0.f ? v : NEG_SLOPE * v;
}

// CDNA5 async global->LDS copy path (ASYNCcnt-tracked), guarded so the file
// still compiles if this toolchain doesn't declare the builtins.
// Probe-learned signature: (v4i addrspace(1)*, v4i addrspace(3)*, imm offset, imm cpol)
#if defined(__has_builtin)
#if __has_builtin(__builtin_amdgcn_global_load_async_to_lds_b128) && \
    __has_builtin(__builtin_amdgcn_s_wait_asynccnt)
#define USE_ASYNC_LDS 1
#endif
#endif

typedef __attribute__((address_space(1))) v4i GV4;  // global v4i
typedef __attribute__((address_space(3))) v4i LV4;  // LDS v4i

// ---------------------------------------------------------------------------
// Kernel 1: kNN (top-20 by negative squared distance). One batch's points
// (48KB) live in LDS; each thread owns one row and keeps a sorted top-20 in
// registers (fully unrolled insertion, guarded by current-min compare).
// ---------------------------------------------------------------------------
__global__ __launch_bounds__(128)
void knn_kernel(const float* __restrict__ x, int* __restrict__ idx) {
    const int b   = blockIdx.y;
    const int row = blockIdx.x * 128 + threadIdx.x;
    __shared__ float sx[NPTS], sy[NPTS], sz[NPTS];
    const float* xb = x + (size_t)b * NPTS * 3;
    for (int i = threadIdx.x; i < NPTS; i += 128) {
        sx[i] = xb[i * 3 + 0];
        sy[i] = xb[i * 3 + 1];
        sz[i] = xb[i * 3 + 2];
    }
    __syncthreads();

    const float cx = sx[row], cy = sy[row], cz = sz[row];
    float bd[KNN];
    int   bi[KNN];
#pragma unroll
    for (int s = 0; s < KNN; ++s) { bd[s] = -1e30f; bi[s] = 0; }

    for (int j = 0; j < NPTS; ++j) {
        const float dx = sx[j] - cx;
        const float dy = sy[j] - cy;
        const float dz = sz[j] - cz;
        const float d  = -(dx * dx + dy * dy + dz * dz);
        if (d > bd[KNN - 1]) {
            float vd = d; int vi = j;
#pragma unroll
            for (int s = 0; s < KNN; ++s) {
                if (vd > bd[s]) {
                    float td = bd[s]; bd[s] = vd; vd = td;
                    int   ti = bi[s]; bi[s] = vi; vi = ti;
                }
            }
        }
    }
    int* op = idx + ((size_t)b * NPTS + row) * KNN;
#pragma unroll
    for (int s = 0; s < KNN; ++s) op[s] = bi[s];
}

// ---------------------------------------------------------------------------
// Kernel 2: EdgeConv layer 0 (6 -> 64, affine + lrelu, max over k neighbors).
// One 64-thread block per point; neighbor coords staged in LDS; the "center"
// half of the edge feature is hoisted out of the neighbor loop.
// Output h0: [B][64][N] f16 (channel-major to feed WMMA B-tiles).
// ---------------------------------------------------------------------------
__global__ __launch_bounds__(64)
void edgeconv0_kernel(const float* __restrict__ x, const int* __restrict__ idx,
                      const float* __restrict__ W0, const float* __restrict__ b0,
                      const float* __restrict__ s0, const float* __restrict__ t0,
                      _Float16* __restrict__ h0) {
    const int b = blockIdx.y;
    const int n = blockIdx.x;
    const int o = threadIdx.x;  // output channel 0..63

    __shared__ float snb[KNN][3];
    __shared__ float sc[3];
    const float* xb = x + (size_t)b * NPTS * 3;
    if (o < 3) sc[o] = xb[(size_t)n * 3 + o];
    if (o < KNN) {
        const int j = idx[((size_t)b * NPTS + n) * KNN + o];
        snb[o][0] = xb[(size_t)j * 3 + 0];
        snb[o][1] = xb[(size_t)j * 3 + 1];
        snb[o][2] = xb[(size_t)j * 3 + 2];
    }
    __syncthreads();

    const float w0 = W0[o * 6 + 0], w1 = W0[o * 6 + 1], w2 = W0[o * 6 + 2];
    const float w3 = W0[o * 6 + 3], w4 = W0[o * 6 + 4], w5 = W0[o * 6 + 5];
    const float ss = s0[o], tt = t0[o];
    const float cpart = w3 * sc[0] + w4 * sc[1] + w5 * sc[2] + b0[o];

    float m = -1e30f;
#pragma unroll 4
    for (int j = 0; j < KNN; ++j) {
        const float pre = w0 * (snb[j][0] - sc[0]) +
                          w1 * (snb[j][1] - sc[1]) +
                          w2 * (snb[j][2] - sc[2]) + cpart;
        m = fmaxf(m, lrelu_f(pre * ss + tt));
    }
    h0[((size_t)b * 64 + o) * NPTS + n] = (_Float16)m;
}

// ---------------------------------------------------------------------------
// Kernel 3: 1x1 conv + affine + lrelu as WMMA GEMM: out[o,n] = act over
// W[COUT,CIN] * H[CIN,N].
//   A = W row tiles (f16, converted in LDS),
//   B = activation tile sH[c][n] staged via GLOBAL_LOAD_ASYNC_TO_LDS_B128
//       (row = 128 contiguous bytes; no VGPR round-trip), consumed with
//       strided ds reads in this code's own K order, same as A
//       (lane-half holds 16 consecutive K values).
//   f16 inputs, f32 accumulate via v_wmma_f32_16x16x32_f16.
// ---------------------------------------------------------------------------
template <int CIN, int COUT>
__global__ __launch_bounds__(256)
void pconv_wmma_kernel(const _Float16* __restrict__ Hin,
                       const float* __restrict__ W, const float* __restrict__ bias,
                       const float* __restrict__ s, const float* __restrict__ t,
                       _Float16* __restrict__ Hout) {
    constexpr int TILE_N = 64;
    const int b  = blockIdx.y;
    const int n0 = blockIdx.x * TILE_N;

    __shared__ _Float16 sW[COUT * CIN];       // [o][c] row-major
    __shared__ _Float16 sH[CIN * TILE_N];     // [c][n] row-major (global layout)

    const int tid = threadIdx.x;
    for (int i = tid; i < COUT * CIN; i += 256) sW[i] = (_Float16)W[i];

    const _Float16* src = Hin + (size_t)b * CIN * NPTS;
#if USE_ASYNC_LDS
    // 16-byte async transfers: 8 chunks per CIN row of the tile.
    constexpr int CHUNKS = CIN * TILE_N / 8;  // 8 halves per chunk
    for (int i = tid; i < CHUNKS; i += 256) {
        const int c     = i >> 3;
        const int chunk = i & 7;
        const _Float16* g = src + (size_t)c * NPTS + n0 + chunk * 8;
        _Float16*       l = &sH[c * TILE_N + chunk * 8];
        __builtin_amdgcn_global_load_async_to_lds_b128((GV4*)g, (LV4*)l, 0, 0);
    }
    __builtin_amdgcn_s_wait_asynccnt(0);
#else
    for (int i = tid; i < CIN * TILE_N; i += 256)
        sH[i] = src[(size_t)(i >> 6) * NPTS + n0 + (i & (TILE_N - 1))];
#endif
    __syncthreads();

    const int wave = tid >> 5, lane = tid & 31;
    const int lrow = lane & 15;
    const int half = lane >> 4;          // 0 or 1
    const int kgA  = half << 4;          // A: 16 consecutive K per lane-half
    const int kgB  = half << 4;          // B: 16 consecutive K per lane-half

    constexpr int TM = COUT / 16;
    constexpr int TN = TILE_N / 16;      // 4

    for (int tile = wave; tile < TM * TN; tile += 8) {
        const int tm = tile / TN;
        const int tn = tile % TN;
        v8f acc = {};
#pragma unroll
        for (int k0 = 0; k0 < CIN; k0 += 32) {
            // A fragment: W[tm*16+lrow][k0 + kgA .. k0 + kgA+15]
            const _Float16* wrow = &sW[(tm * 16 + lrow) * CIN + k0];
            v8h alo = *(const v8h*)(wrow + kgA);
            v8h ahi = *(const v8h*)(wrow + kgA + 8);
            v16h a  = __builtin_shufflevector(alo, ahi, 0, 1, 2, 3, 4, 5, 6, 7,
                                              8, 9, 10, 11, 12, 13, 14, 15);
            // B fragment: column n = tn*16+lrow, K = k0+kgB .. k0+kgB+15
            const _Float16* hcol = &sH[(k0 + kgB) * TILE_N + tn * 16 + lrow];
            v16h bm;
#pragma unroll
            for (int i = 0; i < 16; ++i) bm[i] = hcol[i * TILE_N];
            acc = __builtin_amdgcn_wmma_f32_16x16x32_f16(
                false, a, false, bm, (short)0, acc, false, false);
        }
        // D: lane holds column n; VGPR r holds row r (lanes 0-15) / 8+r (16-31)
        const int ncol = n0 + tn * 16 + lrow;
        _Float16* dst  = Hout + (size_t)b * COUT * NPTS + ncol;
        const int mbase = tm * 16 + (half << 3);
#pragma unroll
        for (int r = 0; r < 8; ++r) {
            const int m = mbase + r;
            float v = (acc[r] + bias[m]) * s[m] + t[m];
            dst[(size_t)m * NPTS] = (_Float16)lrelu_f(v);
        }
    }
}

// ---------------------------------------------------------------------------
// Kernel 4: max over N per (b, channel).
// ---------------------------------------------------------------------------
__global__ __launch_bounds__(256)
void maxpool_kernel(const _Float16* __restrict__ H, float* __restrict__ out) {
    const int b = blockIdx.y;
    const int o = blockIdx.x;  // 0..127
    const _Float16* src = H + ((size_t)b * 128 + o) * NPTS;
    float m = -1e30f;
    for (int i = threadIdx.x; i < NPTS; i += 256) m = fmaxf(m, (float)src[i]);
    __shared__ float red[256];
    red[threadIdx.x] = m;
    __syncthreads();
    for (int st = 128; st > 0; st >>= 1) {
        if (threadIdx.x < st)
            red[threadIdx.x] = fmaxf(red[threadIdx.x], red[threadIdx.x + st]);
        __syncthreads();
    }
    if (threadIdx.x == 0) out[b * 128 + o] = red[0];
}

// ---------------------------------------------------------------------------
// Kernel 5/6: head. 8x512 and 8x1024 dot products — pure VALU, trivial work.
// ---------------------------------------------------------------------------
__global__ __launch_bounds__(256)
void head1_kernel(const float* __restrict__ pooled, const float* __restrict__ W4,
                  const float* __restrict__ b4, const float* __restrict__ s4,
                  const float* __restrict__ t4, float* __restrict__ fc) {
    const int id = blockIdx.x * 256 + threadIdx.x;  // 4096 = 8*512
    const int b = id >> 9, o = id & 511;
    const float* h = pooled + b * 128;
    const float* w = W4 + o * 128;
    float acc = 0.f;
#pragma unroll 8
    for (int c = 0; c < 128; ++c) acc += h[c] * w[c];
    fc[id] = lrelu_f((acc + b4[o]) * s4[o] + t4[o]);
}

__global__ __launch_bounds__(256)
void head2_kernel(const float* __restrict__ fc, const float* __restrict__ W5,
                  const float* __restrict__ b5, float* __restrict__ out) {
    const int id = blockIdx.x * 256 + threadIdx.x;  // 8192 = 8*1024
    const int b = id >> 10, o = id & 1023;
    const float* h = fc + b * 512;
    const float* w = W5 + o * 512;
    float acc = 0.f;
#pragma unroll 8
    for (int c = 0; c < 512; ++c) acc += h[c] * w[c];
    out[id] = acc + b5[o];
}

// ---------------------------------------------------------------------------
extern "C" void kernel_launch(void* const* d_in, const int* in_sizes, int n_in,
                              void* d_out, int out_size, void* d_ws, size_t ws_size,
                              hipStream_t stream) {
    (void)in_sizes; (void)n_in; (void)out_size; (void)ws_size;

    const float* x  = (const float*)d_in[0];
    const float* W0 = (const float*)d_in[1];
    const float* b0 = (const float*)d_in[2];
    const float* s0 = (const float*)d_in[3];
    const float* t0 = (const float*)d_in[4];
    const float* W1 = (const float*)d_in[5];
    const float* b1 = (const float*)d_in[6];
    const float* s1 = (const float*)d_in[7];
    const float* t1 = (const float*)d_in[8];
    const float* W2 = (const float*)d_in[9];
    const float* b2 = (const float*)d_in[10];
    const float* s2 = (const float*)d_in[11];
    const float* t2 = (const float*)d_in[12];
    const float* W3 = (const float*)d_in[13];
    const float* b3 = (const float*)d_in[14];
    const float* s3 = (const float*)d_in[15];
    const float* t3 = (const float*)d_in[16];
    const float* W4 = (const float*)d_in[17];
    const float* b4 = (const float*)d_in[18];
    const float* s4 = (const float*)d_in[19];
    const float* t4 = (const float*)d_in[20];
    const float* W5 = (const float*)d_in[21];
    const float* b5 = (const float*)d_in[22];
    float* out = (float*)d_out;

    // Workspace layout (all offsets 256B-aligned by construction)
    char* ws = (char*)d_ws;
    size_t off = 0;
    int*       d_idx = (int*)(ws + off);       off += (size_t)NB * NPTS * KNN * sizeof(int);      // 2.62 MB
    _Float16*  h0    = (_Float16*)(ws + off);  off += (size_t)NB * 64  * NPTS * sizeof(_Float16); // 4 MB
    _Float16*  h1    = (_Float16*)(ws + off);  off += (size_t)NB * 64  * NPTS * sizeof(_Float16); // 4 MB
    _Float16*  h2    = (_Float16*)(ws + off);  off += (size_t)NB * 128 * NPTS * sizeof(_Float16); // 8 MB
    _Float16*  h3    = (_Float16*)(ws + off);  off += (size_t)NB * 128 * NPTS * sizeof(_Float16); // 8 MB
    float*     pool  = (float*)(ws + off);     off += (size_t)NB * 128 * sizeof(float);
    float*     fc    = (float*)(ws + off);     off += (size_t)NB * 512 * sizeof(float);

    // 1) kNN indices
    knn_kernel<<<dim3(NPTS / 128, NB), 128, 0, stream>>>(x, d_idx);
    // 2) EdgeConv 6->64 + max over k
    edgeconv0_kernel<<<dim3(NPTS, NB), 64, 0, stream>>>(x, d_idx, W0, b0, s0, t0, h0);
    // 3) WMMA 1x1-conv stack
    pconv_wmma_kernel<64, 64><<<dim3(NPTS / 64, NB), 256, 0, stream>>>(h0, W1, b1, s1, t1, h1);
    pconv_wmma_kernel<64, 128><<<dim3(NPTS / 64, NB), 256, 0, stream>>>(h1, W2, b2, s2, t2, h2);
    pconv_wmma_kernel<128, 128><<<dim3(NPTS / 64, NB), 256, 0, stream>>>(h2, W3, b3, s3, t3, h3);
    // 4) global max pool over points
    maxpool_kernel<<<dim3(128, NB), 256, 0, stream>>>(h3, pool);
    // 5) head
    head1_kernel<<<dim3(16), 256, 0, stream>>>(pool, W4, b4, s4, t4, fc);
    head2_kernel<<<dim3(32), 256, 0, stream>>>(fc, W5, b5, out);
}
